// TCGE_17832704213284
// MI455X (gfx1250) — compile-verified, matrix operand order checked by reading, not run on hardware
//
#include <hip/hip_runtime.h>
#include <hip/hip_bf16.h>
#include <math.h>
#include <stdint.h>

// ---------------------------------------------------------------------------
// MI455X (gfx1250) implementation.
// Dead-code eliminated TCGE forward (gather index = lengths-1 < 256 => conv
// branches + 468/469 MLP rows never reach the output).
// GEMMs: v_wmma_f32_16x16x32_bf16, operands staged LDS-side with double-
// buffered global_load_async_to_lds_b128 (ASYNCcnt) -- no VGPR staging, no
// per-tile convert VALU.  All matrix operands pre-converted to bf16 once.
// ---------------------------------------------------------------------------

#define BSZ 128
#define SEQ 256
#define DIM 512
#define DKH 256   // head dim (2 heads)

typedef __attribute__((ext_vector_type(16))) __bf16 v16bf;
typedef __attribute__((ext_vector_type(8)))  __bf16 v8bf;
typedef __attribute__((ext_vector_type(8)))  float  v8f;
typedef __attribute__((ext_vector_type(4)))  float  v4f;

__device__ __forceinline__ __bf16 f2bf(float f) {
  unsigned u = __builtin_bit_cast(unsigned, f);
  unsigned r = u + 0x7FFFu + ((u >> 16) & 1u);        // round-to-nearest-even
  unsigned short h = (unsigned short)(r >> 16);
  return __builtin_bit_cast(__bf16, h);
}
__device__ __forceinline__ float bf2f(__bf16 h) {
  unsigned short u = __builtin_bit_cast(unsigned short, h);
  return __builtin_bit_cast(float, (unsigned)u << 16);
}

// gfx1250 async DMA: global -> LDS, 16 bytes per lane, tracked on ASYNCcnt.
__device__ __forceinline__ void async_b128(unsigned lds_off, const void* gaddr) {
  asm volatile("global_load_async_to_lds_b128 %0, %1, off"
               :: "v"(lds_off), "v"(gaddr) : "memory");
}
__device__ __forceinline__ void wait_async0() {
  asm volatile("s_wait_asynccnt 0x0" ::: "memory");
}
__device__ __forceinline__ unsigned lds_off_of(const void* p) {
  return (unsigned)(uintptr_t)p;     // flat LDS addr: low 32 bits = LDS offset
}

// ---------------------------------------------------------------------------
// one-shot f32 -> bf16 conversion (n multiple of 2048)
// ---------------------------------------------------------------------------
__global__ __launch_bounds__(256)
void cvt_f32_bf16(const float* __restrict__ s, __bf16* __restrict__ d) {
  const int i = (blockIdx.x * 256 + threadIdx.x) * 8;
  v4f lo = *(const v4f*)(s + i);
  v4f hi = *(const v4f*)(s + i + 4);
  v8bf pk;
  pk[0] = f2bf(lo.x); pk[1] = f2bf(lo.y); pk[2] = f2bf(lo.z); pk[3] = f2bf(lo.w);
  pk[4] = f2bf(hi.x); pk[5] = f2bf(hi.y); pk[6] = f2bf(hi.z); pk[7] = f2bf(hi.w);
  *(v8bf*)(d + i) = pk;
}

// ---------------------------------------------------------------------------
// C[M,N] = epi( A[M,K] @ W[N,K]^T + bias ),  bf16 A/W/Cin/C, f32 accumulate.
// mode 0: C = t     mode 1: C = Cin * t     mode 2: C = Cin * sigmoid(t)
// Block: 256 threads = 8 waves; tile 128x64, BK=32, double-buffered async DMA.
// Wave w -> rows [16w,16w+16) x 64 cols = 4 x (16x16 f32 accum).
// ---------------------------------------------------------------------------
__global__ __launch_bounds__(256)
void gemm_wmma_bf16(const __bf16* __restrict__ A, int lda,
                    const __bf16* __restrict__ W,
                    const float* __restrict__ bias,
                    const __bf16* __restrict__ Cin, int ldci,
                    __bf16* __restrict__ C, int ldc,
                    int K, int mode)
{
  __shared__ __attribute__((aligned(32))) __bf16 sA[2][128 * 32];
  __shared__ __attribute__((aligned(32))) __bf16 sB[2][64 * 32];

  const int tid  = threadIdx.x;
  const int wave = tid >> 5;
  const int lane = tid & 31;
  const int m0   = blockIdx.x * 128;
  const int n0   = blockIdx.y * 64;

  const int lrow = lane & 15;
  const int half = lane >> 4;       // 0 -> K 0..15 side, 1 -> K 16..31 side
  const int koff = half << 4;

  v8f acc[4] = {};

  const unsigned sA0 = lds_off_of(&sA[0][0]);
  const unsigned sA1 = lds_off_of(&sA[1][0]);
  const unsigned sB0 = lds_off_of(&sB[0][0]);
  const unsigned sB1 = lds_off_of(&sB[1][0]);

  // stage tile kt into buffer buf: A 128x32 (2 x 16B/lane), B 64x32 (1 x 16B)
  auto stage = [&](int kt, int buf) {
    const int k0 = kt << 5;
    const unsigned aBase = buf ? sA1 : sA0;
    const unsigned bBase = buf ? sB1 : sB0;
    #pragma unroll
    for (int q = 0; q < 2; ++q) {
      const int c   = tid * 2 + q;
      const int r   = c >> 2;
      const int col = (c & 3) << 3;
      async_b128(aBase + (unsigned)(r * 32 + col) * 2,
                 A + (size_t)(m0 + r) * lda + k0 + col);
    }
    {
      const int r   = tid >> 2;
      const int col = (tid & 3) << 3;
      async_b128(bBase + (unsigned)(r * 32 + col) * 2,
                 W + (size_t)(n0 + r) * K + k0 + col);
    }
  };

  const int nk = K >> 5;
  stage(0, 0);
  wait_async0();
  __syncthreads();

  for (int kt = 0; kt < nk; ++kt) {
    const int cur = kt & 1;
    if (kt + 1 < nk) stage(kt + 1, cur ^ 1);   // DMA overlaps WMMA below

    const v16bf af = *(const v16bf*)&sA[cur][(wave * 16 + lrow) * 32 + koff];
    #pragma unroll
    for (int j = 0; j < 4; ++j) {
      const v16bf bf = *(const v16bf*)&sB[cur][(j * 16 + lrow) * 32 + koff];
      acc[j] = __builtin_amdgcn_wmma_f32_16x16x32_bf16(
          false, af, false, bf, (short)0, acc[j], false, false);
    }
    wait_async0();        // own-wave DMA for next tile done
    __syncthreads();      // all waves done writing next buf / reading cur buf
  }

  // ---- epilogue: 16x16 f32 C layout (VGPR r: M = r + 8*half, N = lrow) ----
  const int rbase = m0 + wave * 16 + (half << 3);
  #pragma unroll
  for (int j = 0; j < 4; ++j) {
    const int col = n0 + j * 16 + lrow;
    const float bj = bias[col];
    #pragma unroll
    for (int r = 0; r < 8; ++r) {
      const size_t row = (size_t)(rbase + r);
      float t = acc[j][r] + bj;
      if (mode == 1) {
        t = bf2f(Cin[row * ldci + col]) * t;
      } else if (mode == 2) {
        t = bf2f(Cin[row * ldci + col]) * (1.0f / (1.0f + __expf(-t)));
      }
      C[row * ldc + col] = f2bf(t);
    }
  }
}

// ---------------------------------------------------------------------------
// Per-batch single-row attention + BN1 + residual (bf16 qg/kg/v, f32 math).
// ---------------------------------------------------------------------------
__global__ __launch_bounds__(256)
void attn_row_kernel(const __bf16* __restrict__ QG, const __bf16* __restrict__ KG,
                     const __bf16* __restrict__ V,  const float* __restrict__ cap,
                     const int* __restrict__ lengths,
                     const float* __restrict__ g1, const float* __restrict__ b1,
                     const float* __restrict__ m1, const float* __restrict__ v1,
                     float* __restrict__ ga_row)
{
  const int b   = blockIdx.x;
  const int tid = threadIdx.x;
  const int i   = lengths[b] - 1;

  __shared__ float s_q[DKH];
  __shared__ float s_p[SEQ];
  __shared__ float red[256];

  const size_t base = (size_t)b * SEQ * DIM;

  for (int h = 0; h < 2; ++h) {
    const int ho = h * DKH;
    s_q[tid] = bf2f(QG[base + (size_t)i * DIM + ho + tid]);
    __syncthreads();

    // score for key position s = tid
    const v8bf* kr8 = (const v8bf*)(KG + base + (size_t)tid * DIM + ho);
    float dot = 0.f;
    #pragma unroll 4
    for (int d8 = 0; d8 < DKH / 8; ++d8) {
      const v8bf kv = kr8[d8];
      #pragma unroll
      for (int e = 0; e < 8; ++e) dot += s_q[d8 * 8 + e] * bf2f(kv[e]);
    }
    const float sc = floorf(dot * 0.0625f);   // floor(./sqrt(256))

    red[tid] = sc; __syncthreads();
    for (int off = 128; off > 0; off >>= 1) {
      if (tid < off) red[tid] = fmaxf(red[tid], red[tid + off]);
      __syncthreads();
    }
    const float mx = red[0]; __syncthreads();

    const float p = __expf(sc - mx);
    red[tid] = p; __syncthreads();
    for (int off = 128; off > 0; off >>= 1) {
      if (tid < off) red[tid] += red[tid + off];
      __syncthreads();
    }
    const float inv = 1.0f / red[0]; __syncthreads();
    s_p[tid] = p * inv; __syncthreads();

    // out[d=tid] = sum_s p[s] * V[b,s,ho+d]  (coalesced across threads)
    float acc = 0.f;
    const __bf16* vp = V + base + ho + tid;
    #pragma unroll 4
    for (int s = 0; s < SEQ; ++s) acc += s_p[s] * bf2f(vp[(size_t)s * DIM]);

    const int c  = ho + tid;
    const float xn = (acc - m1[c]) * (g1[c] * rsqrtf(v1[c] + 1e-5f)) + b1[c];
    ga_row[(size_t)b * DIM + c] = cap[base + (size_t)i * DIM + c] + xn;
    __syncthreads();
  }
}

// ---------------------------------------------------------------------------
// Per-batch row MLP (512 -> 1024 relu -> 512) + residual + l2 normalize.
// ---------------------------------------------------------------------------
__global__ __launch_bounds__(256)
void mlp_norm_kernel(const float* __restrict__ ga_row,
                     const float* __restrict__ Wm1, const float* __restrict__ bm1,
                     const float* __restrict__ Wm2, const float* __restrict__ bm2,
                     float* __restrict__ out)
{
  const int b   = blockIdx.x;
  const int tid = threadIdx.x;
  __shared__ float s_row[DIM];
  __shared__ float s_hid[2 * DIM];
  __shared__ float s_o[DIM];
  __shared__ float red[256];

  s_row[tid]       = ga_row[(size_t)b * DIM + tid];
  s_row[tid + 256] = ga_row[(size_t)b * DIM + tid + 256];
  __syncthreads();

  #pragma unroll
  for (int jj = 0; jj < 4; ++jj) {
    const int j = jj * 256 + tid;
    const float* w = Wm1 + (size_t)j * DIM;
    float acc = bm1[j];
    #pragma unroll 8
    for (int d = 0; d < DIM; ++d) acc += s_row[d] * w[d];
    s_hid[j] = fmaxf(acc, 0.f);
  }
  __syncthreads();

  float sq = 0.f;
  #pragma unroll
  for (int cc = 0; cc < 2; ++cc) {
    const int c = cc * 256 + tid;
    const float* w = Wm2 + (size_t)c * (2 * DIM);
    float acc = bm2[c];
    #pragma unroll 8
    for (int d = 0; d < 2 * DIM; ++d) acc += s_hid[d] * w[d];
    acc += s_row[c];
    s_o[c] = acc;
    sq += acc * acc;
  }
  red[tid] = sq; __syncthreads();
  for (int off = 128; off > 0; off >>= 1) {
    if (tid < off) red[tid] += red[tid + off];
    __syncthreads();
  }
  const float inv = 1.0f / (sqrtf(red[0]) + 1e-8f);
  out[(size_t)b * DIM + tid]       = s_o[tid] * inv;
  out[(size_t)b * DIM + tid + 256] = s_o[tid + 256] * inv;
}

// ---------------------------------------------------------------------------
extern "C" void kernel_launch(void* const* d_in, const int* in_sizes, int n_in,
                              void* d_out, int out_size, void* d_ws, size_t ws_size,
                              hipStream_t stream) {
  (void)in_sizes; (void)n_in; (void)out_size; (void)ws_size;

  const float* cap     = (const float*)d_in[0];
  const int*   lengths = (const int*)d_in[1];
  const float* Wq  = (const float*)d_in[2];
  const float* Wk  = (const float*)d_in[3];
  const float* Wv  = (const float*)d_in[4];
  const float* bq  = (const float*)d_in[5];
  const float* bk  = (const float*)d_in[6];
  const float* bv  = (const float*)d_in[7];
  const float* Wfq = (const float*)d_in[8];
  const float* bfq = (const float*)d_in[9];
  const float* Wfk = (const float*)d_in[10];
  const float* bfk = (const float*)d_in[11];
  const float* Wfg = (const float*)d_in[12];
  const float* bfg = (const float*)d_in[13];
  const float* bn1g = (const float*)d_in[14];
  const float* bn1b = (const float*)d_in[15];
  const float* bn1m = (const float*)d_in[16];
  const float* bn1v = (const float*)d_in[17];
  // d_in[18..29]: bn2/bn3/conv weights -- provably dead for the output
  const float* Wm1 = (const float*)d_in[30];
  const float* bm1 = (const float*)d_in[31];
  const float* Wm2 = (const float*)d_in[32];
  const float* bm2 = (const float*)d_in[33];

  const size_t NT = (size_t)BSZ * SEQ;   // 32768 tokens

  // --- bf16 workspace carve-up ---
  __bf16* capb = (__bf16*)d_ws;          // [NT,512]
  __bf16* Qb   = capb + NT * DIM;        // [NT,512] q -> qg in place
  __bf16* Kbb  = Qb   + NT * DIM;        // [NT,512] k -> kg in place
  __bf16* Vbb  = Kbb  + NT * DIM;        // [NT,512]
  __bf16* Gb   = Vbb  + NT * DIM;        // [NT,512] FQ -> FQ*FK
  __bf16* Wqb  = Gb   + NT * DIM;
  __bf16* Wkb  = Wqb  + DIM * DIM;
  __bf16* Wvb  = Wkb  + DIM * DIM;
  __bf16* Wfqb = Wvb  + DIM * DIM;
  __bf16* Wfkb = Wfqb + DKH * DKH;
  __bf16* Wfgb = Wfkb + DKH * DKH;       // [512,256]
  float*  ga   = (float*)(Wfgb + DIM * DKH);   // [128,512] gathered rows

  dim3 blk(256);

  // --- one-shot bf16 conversion of all matrix operands ---
  cvt_f32_bf16<<<dim3((unsigned)(NT * DIM / 2048)), blk, 0, stream>>>(cap, capb);
  cvt_f32_bf16<<<dim3(DIM * DIM / 2048), blk, 0, stream>>>(Wq, Wqb);
  cvt_f32_bf16<<<dim3(DIM * DIM / 2048), blk, 0, stream>>>(Wk, Wkb);
  cvt_f32_bf16<<<dim3(DIM * DIM / 2048), blk, 0, stream>>>(Wv, Wvb);
  cvt_f32_bf16<<<dim3(DKH * DKH / 2048), blk, 0, stream>>>(Wfq, Wfqb);
  cvt_f32_bf16<<<dim3(DKH * DKH / 2048), blk, 0, stream>>>(Wfk, Wfkb);
  cvt_f32_bf16<<<dim3(DIM * DKH / 2048), blk, 0, stream>>>(Wfg, Wfgb);

  dim3 g512((unsigned)(NT / 128), 8);    // N=512
  dim3 g256((unsigned)(NT / 128), 4);    // N=256

  // Q,K,V = cap @ {Wq,Wk,Wv}^T + b   (K=512)
  gemm_wmma_bf16<<<g512, blk, 0, stream>>>(capb, DIM, Wqb, bq, nullptr, 0, Qb,  DIM, DIM, 0);
  gemm_wmma_bf16<<<g512, blk, 0, stream>>>(capb, DIM, Wkb, bk, nullptr, 0, Kbb, DIM, DIM, 0);
  gemm_wmma_bf16<<<g512, blk, 0, stream>>>(capb, DIM, Wvb, bv, nullptr, 0, Vbb, DIM, DIM, 0);

  // per head: FQ = q@Wfq^T+bfq  (into G)
  for (int h = 0; h < 2; ++h) {
    const int o = h * DKH;
    gemm_wmma_bf16<<<g256, blk, 0, stream>>>(Qb + o, DIM, Wfqb, bfq, nullptr, 0,
                                             Gb + o, DIM, DKH, 0);
  }
  // per head: G = FQ * (k@Wfk^T+bfk)
  for (int h = 0; h < 2; ++h) {
    const int o = h * DKH;
    gemm_wmma_bf16<<<g256, blk, 0, stream>>>(Kbb + o, DIM, Wfkb, bfk, Gb + o, DIM,
                                             Gb + o, DIM, DKH, 1);
  }
  // per head: qg = q * sigmoid(G@Wfg[0:256]^T+bfg[:256])    (Q in place)
  //           kg = k * sigmoid(G@Wfg[256:512]^T+bfg[256:])  (K in place)
  for (int h = 0; h < 2; ++h) {
    const int o = h * DKH;
    gemm_wmma_bf16<<<g256, blk, 0, stream>>>(Gb + o, DIM, Wfgb, bfg, Qb + o, DIM,
                                             Qb + o, DIM, DKH, 2);
    gemm_wmma_bf16<<<g256, blk, 0, stream>>>(Gb + o, DIM, Wfgb + DKH * DKH, bfg + DKH,
                                             Kbb + o, DIM, Kbb + o, DIM, DKH, 2);
  }

  attn_row_kernel<<<dim3(BSZ), blk, 0, stream>>>(Qb, Kbb, Vbb, cap, lengths,
                                                 bn1g, bn1b, bn1m, bn1v, ga);
  mlp_norm_kernel<<<dim3(BSZ), blk, 0, stream>>>(ga, Wm1, bm1, Wm2, bm2,
                                                 (float*)d_out);
}
